// ColSentenceModel_64879775974007
// MI455X (gfx1250) — matrix operand order repaired and checked by reading, not compiled, not run on hardware
//
#include <hip/hip_runtime.h>

typedef _Float16 v16h __attribute__((ext_vector_type(16)));
typedef _Float16 h8   __attribute__((ext_vector_type(8)));
typedef _Float16 h4   __attribute__((ext_vector_type(4)));
typedef float    v8f  __attribute__((ext_vector_type(8)));
typedef float    f4   __attribute__((ext_vector_type(4)));

#define ND 128
#define LD 128
#define EDIM 128
#define NQ 128
#define LQ 32
#define LDS_STRIDE 136   // 128 + 8 halves pad -> 16B bank shift per row, conflict-free B128 reads

union Frag { v16h v; h8 h[2]; };

__global__ __launch_bounds__(256)
void colbert_maxsim_kernel(const float* __restrict__ doc,
                           const float* __restrict__ query,
                           float* __restrict__ out) {
  __shared__ _Float16 docL[LD * LDS_STRIDE];   // 34,816 bytes

  const int tid  = threadIdx.x;
  const int lane = tid & 31;
  const int wave = tid >> 5;
  const int d = blockIdx.x;
  const int q = blockIdx.y * 8 + wave;

  // ---- cooperative: convert doc[d] fp32 -> f16 into LDS (row-major, padded) ----
  const f4* docG4 = (const f4*)(doc + (size_t)d * LD * EDIM);
#pragma unroll
  for (int it = 0; it < (LD * EDIM) / (256 * 4); ++it) {
    const int flat = (it * 256 + tid) * 4;
    const int row = flat >> 7;
    const int col = flat & 127;
    f4 v = docG4[it * 256 + tid];
    h4 hv = { (_Float16)v.x, (_Float16)v.y, (_Float16)v.z, (_Float16)v.w };
    *(h4*)(docL + row * LDS_STRIDE + col) = hv;     // ds_store_b64
  }

  // ---- per-wave: preload this query's B fragments from global (fp32 -> f16) ----
  // B layout (32x16 f16): lanes 0-15, col N=lane, halves 0..15 = K kb..kb+15;
  // lanes 16-31, col N=lane-16, halves 0..15 = K kb+16..kb+31.
  const float* qG  = query + (size_t)q * EDIM * LQ;
  const int ncol   = lane & 15;
  const int khalf  = (lane >> 4) * 16;
  Frag b[4][2];
#pragma unroll
  for (int ks = 0; ks < 4; ++ks) {
#pragma unroll
    for (int nt = 0; nt < 2; ++nt) {
      const int c = nt * 16 + ncol;
#pragma unroll
      for (int j = 0; j < 16; ++j) {
        const int k = ks * 32 + khalf + j;
        b[ks][nt].v[j] = (_Float16)qG[k * LQ + c];
      }
    }
  }

  __syncthreads();

  // ---- GEMM over 8 M-tiles (doc tokens t), with running column-max ----
  // A layout (16x32 f16): lane&15 = row M; lanes<16 hold K kb..kb+15,
  // lanes>=16 hold K kb+16..kb+31 (kb = ks*32).
  const int arow  = lane & 15;
  const int akoff = (lane >> 4) * 16;
  float cm0 = -3.402823466e38f, cm1 = -3.402823466e38f;
#pragma unroll
  for (int mt = 0; mt < 8; ++mt) {
    v8f acc0 = {0.f, 0.f, 0.f, 0.f, 0.f, 0.f, 0.f, 0.f};
    v8f acc1 = acc0;
    const _Float16* arowp = docL + (mt * 16 + arow) * LDS_STRIDE;
#pragma unroll
    for (int ks = 0; ks < 4; ++ks) {
      Frag a;
      const h8* ap = (const h8*)(arowp + ks * 32 + akoff);
      a.h[0] = ap[0];   // K = kb+akoff   .. kb+akoff+7   (ds_load_b128)
      a.h[1] = ap[1];   // K = kb+akoff+8 .. kb+akoff+15  (ds_load_b128)
      acc0 = __builtin_amdgcn_wmma_f32_16x16x32_f16(false, a.v, false, b[ks][0].v,
                                                    (short)0, acc0, false, false);
      acc1 = __builtin_amdgcn_wmma_f32_16x16x32_f16(false, a.v, false, b[ks][1].v,
                                                    (short)0, acc1, false, false);
    }
    // C/D layout: lane<16 holds rows 0..7 of col (lane&15); lane>=16 rows 8..15.
    float m0 = acc0[0], m1 = acc1[0];
#pragma unroll
    for (int r = 1; r < 8; ++r) { m0 = fmaxf(m0, acc0[r]); m1 = fmaxf(m1, acc1[r]); }
    cm0 = fmaxf(cm0, m0);
    cm1 = fmaxf(cm1, m1);
  }

  // combine row halves: rows 0..7 (lane) with rows 8..15 (lane^16)
  cm0 = fmaxf(cm0, __shfl_xor(cm0, 16, 32));
  cm1 = fmaxf(cm1, __shfl_xor(cm1, 16, 32));
  float s = cm0 + cm1;            // contributions of l = ncol and l = ncol+16
#pragma unroll
  for (int off = 8; off >= 1; off >>= 1) s += __shfl_xor(s, off, 32);
  if (lane == 0) out[q * ND + d] = s;
}

extern "C" void kernel_launch(void* const* d_in, const int* in_sizes, int n_in,
                              void* d_out, int out_size, void* d_ws, size_t ws_size,
                              hipStream_t stream) {
  (void)in_sizes; (void)n_in; (void)out_size; (void)d_ws; (void)ws_size;
  const float* doc   = (const float*)d_in[0];   // [128,128,128] f32
  const float* query = (const float*)d_in[1];   // [128,128,32]  f32
  float* out = (float*)d_out;                   // [128,128]     f32
  dim3 grid(ND, NQ / 8);
  colbert_maxsim_kernel<<<grid, dim3(256), 0, stream>>>(doc, query, out);
}
